// TopologicalAutoencoder_9088150798737
// MI455X (gfx1250) — compile-verified, matrix operand order checked by reading, not run on hardware
//
#include <hip/hip_runtime.h>
#include <hip/hip_bf16.h>

// ---------------------------------------------------------------------------
// Topological autoencoder loss on MI455X (gfx1250, wave32, WMMA).
// N=4096 points, D=8192 ambient, LATENT=128.
// Dominant cost: x @ x.T (275 GFLOP) -> split-bf16 WMMA 16x16x32 (3 products
// hi*hi + hi*lo + lo*hi, fp32 accumulate => ~fp32 accuracy, 2.7x fewer matrix
// instructions than the 16x16x4 f32 path, on the wide bf16 pipe).
// x (128MB) fits in the 192MB L2 -> Gram is matrix-pipe bound, not HBM bound.
// Encode/decode GEMMs stay on exact fp32 WMMA and use async global->LDS
// staging (GLOBAL_LOAD_ASYNC_TO_LDS_B128 + s_wait_asynccnt).
// ---------------------------------------------------------------------------

typedef __attribute__((ext_vector_type(2)))  float  v2f;
typedef __attribute__((ext_vector_type(8)))  float  v8f;
typedef __attribute__((ext_vector_type(8)))  __bf16 v8bf;
typedef __attribute__((ext_vector_type(16))) __bf16 v16bf;
typedef int v4i_t __attribute__((vector_size(16)));   // async builtin pointee

#define NPTS   4096
#define DIM    8192
#define LATENT 128

#if defined(__has_builtin)
#  if __has_builtin(__builtin_amdgcn_global_load_async_to_lds_b128)
#    define HAVE_ASYNC_LDS 1
#  endif
#  if __has_builtin(__builtin_amdgcn_s_wait_asynccnt)
#    define HAVE_WAIT_ASYNC 1
#  endif
#endif

// fp32 WMMA: D(16x16,f32) = A(16x4,f32) * B(4x16,f32) + C
__device__ __forceinline__ v8f wmma_f32(v2f a, v2f b, v8f c) {
    return __builtin_amdgcn_wmma_f32_16x16x4_f32(
        false, a, false, b, (short)0, c, false, false);
}

// bf16 WMMA: D(16x16,f32) = A(16x32,bf16) * B(32x16,bf16) + C
__device__ __forceinline__ v8f wmma_bf16(v16bf a, v16bf b, v8f c) {
    return __builtin_amdgcn_wmma_f32_16x16x32_bf16(
        false, a, false, b, (short)0, c, false, false);
}

__device__ __forceinline__ float wave_sum(float s) {
    #pragma unroll
    for (int off = 16; off > 0; off >>= 1) s += __shfl_down(s, off);
    return s;
}

// Async global->LDS staging (gfx1250 GLOBAL_LOAD_ASYNC_TO_LDS_B128) with a
// synchronous fallback.  Toolchain signature (from diagnostic): parameters are
// generic pointers to a 16-byte int vector, plus imm offset and imm cpol.
__device__ __forceinline__ void stage_b128(void* lds, const void* g) {
#if defined(HAVE_ASYNC_LDS)
    __builtin_amdgcn_global_load_async_to_lds_b128((v4i_t*)g, (v4i_t*)lds, 0, 0);
#else
    *(float4*)lds = *(const float4*)g;
#endif
}

__device__ __forceinline__ void stage_fence() {
#if defined(HAVE_ASYNC_LDS)
#  if defined(HAVE_WAIT_ASYNC)
    __builtin_amdgcn_s_wait_asynccnt(0);
#  else
    asm volatile("s_wait_asynccnt 0" ::: "memory");
#  endif
#endif
}

// Split a float into bf16 hi + bf16 lo (residual), packed stores (4 at a time).
__device__ __forceinline__ void split_store(__bf16* hi, __bf16* lo, float4 v) {
    __bf16 h0 = (__bf16)v.x, h1 = (__bf16)v.y, h2 = (__bf16)v.z, h3 = (__bf16)v.w;
    __bf16 l0 = (__bf16)(v.x - (float)h0);
    __bf16 l1 = (__bf16)(v.y - (float)h1);
    __bf16 l2 = (__bf16)(v.z - (float)h2);
    __bf16 l3 = (__bf16)(v.w - (float)h3);
    union P { __bf16 b[4]; uint2 u; } ph, pl;
    ph.b[0] = h0; ph.b[1] = h1; ph.b[2] = h2; ph.b[3] = h3;
    pl.b[0] = l0; pl.b[1] = l1; pl.b[2] = l2; pl.b[3] = l3;
    *(uint2*)hi = ph.u;
    *(uint2*)lo = pl.u;
}

// Assemble the 16-bit WMMA fragment per ISA layout:
// lanes 0-15 hold K = {0..7, 16..23}, lanes 16-31 hold K = {8..15, 24..31}
// (base = (lane>>4)*8).  Two conflict-free b128 LDS reads.
__device__ __forceinline__ v16bf frag16(const __bf16* rowp, int base) {
    v8bf lo = *(const v8bf*)(rowp + base);
    v8bf hi = *(const v8bf*)(rowp + 16 + base);
    return __builtin_shufflevector(lo, hi, 0, 1, 2, 3, 4, 5, 6, 7,
                                   8, 9, 10, 11, 12, 13, 14, 15);
}

// ---------------------------------------------------------------------------
// Row squared norms (fp32 exact): sq[row] = sum_k X[row][k]^2.
// ---------------------------------------------------------------------------
__global__ __launch_bounds__(256)
void rowsq_kernel(const float* __restrict__ X, float* __restrict__ sq, int d) {
    __shared__ float red[8];
    const int row = blockIdx.x;
    const float* p = X + (size_t)row * d;
    float s = 0.f;
    for (int c = threadIdx.x; c < d; c += 256) { float v = p[c]; s += v * v; }
    s = wave_sum(s);
    const int lane = threadIdx.x & 31, wv = threadIdx.x >> 5;
    if (lane == 0) red[wv] = s;
    __syncthreads();
    if (threadIdx.x == 0) {
        float t = 0.f;
        #pragma unroll
        for (int w = 0; w < 8; ++w) t += red[w];
        sq[row] = t;
    }
}

// ---------------------------------------------------------------------------
// Gram + Euclidean distance via split-bf16 WMMA 16x16x32.
// dist[i][j] = sqrt(max(sq[i]+sq[j]-2*<x_i,x_j>,0))
// 64x64 tile / 256-thread block; 8 waves x 2 subtiles; K=64 staged per barrier
// into 4 bf16 LDS tiles (hi/lo for A and B), row stride 72 bf16 (144 B:
// 16B-aligned, conflict-free b128 fragment reads).
// ---------------------------------------------------------------------------
#define BSTR 72

__global__ __launch_bounds__(256)
void gram_dist_kernel(const float* __restrict__ X, const float* __restrict__ sq,
                      float* __restrict__ dist, int n, int d) {
    __shared__ __align__(16) __bf16 Ahi[64 * BSTR];
    __shared__ __align__(16) __bf16 Alo[64 * BSTR];
    __shared__ __align__(16) __bf16 Bhi[64 * BSTR];
    __shared__ __align__(16) __bf16 Blo[64 * BSTR];

    const int ib = blockIdx.y * 64;
    const int jb = blockIdx.x * 64;
    const int tid  = threadIdx.x;
    const int lane = tid & 31;
    const int wv   = tid >> 5;
    const int sr   = wv & 3;            // subtile row (0..3)
    const int sc0  = (wv >> 2) << 1;    // subtile col base (0 or 2)
    const int lr   = lane & 15;
    const int base = (lane >> 4) << 3;  // K sub-offset per 16-bit A/B layout

    v8f acc0 = {}; v8f acc1 = {};

    for (int k0 = 0; k0 < d; k0 += 64) {
        __syncthreads();
        for (int f = tid; f < 1024; f += 256) {        // 64 rows x 16 float4
            const int r = f >> 4, c = (f & 15) << 2;
            float4 av = *(const float4*)&X[(size_t)(ib + r) * d + k0 + c];
            float4 bv = *(const float4*)&X[(size_t)(jb + r) * d + k0 + c];
            split_store(&Ahi[r * BSTR + c], &Alo[r * BSTR + c], av);
            split_store(&Bhi[r * BSTR + c], &Blo[r * BSTR + c], bv);
            if (k0 + 64 < d) {                         // global_prefetch_b8
                __builtin_prefetch(&X[(size_t)(ib + r) * d + k0 + 64 + c], 0, 0);
                __builtin_prefetch(&X[(size_t)(jb + r) * d + k0 + 64 + c], 0, 0);
            }
        }
        __syncthreads();
        const __bf16* ArH = &Ahi[(sr  * 16 + lr) * BSTR];
        const __bf16* ArL = &Alo[(sr  * 16 + lr) * BSTR];
        const __bf16* B0H = &Bhi[(sc0 * 16 + lr) * BSTR];
        const __bf16* B0L = &Blo[(sc0 * 16 + lr) * BSTR];
        #pragma unroll
        for (int kk = 0; kk < 64; kk += 32) {
            v16bf aH  = frag16(ArH + kk, base);
            v16bf aL  = frag16(ArL + kk, base);
            v16bf bH0 = frag16(B0H + kk, base);
            v16bf bL0 = frag16(B0L + kk, base);
            v16bf bH1 = frag16(B0H + 16 * BSTR + kk, base);
            v16bf bL1 = frag16(B0L + 16 * BSTR + kk, base);
            // interleave acc0/acc1 chains to hide WMMA->WMMA RAW latency
            acc0 = wmma_bf16(aH, bH0, acc0);
            acc1 = wmma_bf16(aH, bH1, acc1);
            acc0 = wmma_bf16(aH, bL0, acc0);
            acc1 = wmma_bf16(aH, bL1, acc1);
            acc0 = wmma_bf16(aL, bH0, acc0);
            acc1 = wmma_bf16(aL, bH1, acc1);
        }
    }

    // C layout: VGPR e -> M = e (lanes 0-15) / e+8 (lanes 16-31); N = lane%16
    const int i0 = ib + sr * 16 + ((lane >> 4) << 3);
    const int j0 = jb + sc0 * 16 + lr;
    const int j1 = j0 + 16;
    const float sqj0 = sq[j0], sqj1 = sq[j1];
    #pragma unroll
    for (int e = 0; e < 8; ++e) {
        const int ii = i0 + e;
        const float sqi = sq[ii];
        float d20 = fmaxf(sqi + sqj0 - 2.0f * acc0[e], 0.0f);
        float d21 = fmaxf(sqi + sqj1 - 2.0f * acc1[e], 0.0f);
        dist[(size_t)ii * n + j0] = (d20 > 0.0f) ? sqrtf(d20) : 0.0f;
        dist[(size_t)ii * n + j1] = (d21 > 0.0f) ? sqrtf(d21) : 0.0f;
    }
}

// ---------------------------------------------------------------------------
// Encoder: Z[N][128] = X[N][8192] @ We[8192][128] + be.
// Exact fp32 WMMA 16x16x4; LDS tiles staged with async global->LDS.
// ---------------------------------------------------------------------------
#define GLDS 36

__global__ __launch_bounds__(256)
void encode_kernel(const float* __restrict__ X, const float* __restrict__ We,
                   const float* __restrict__ be, float* __restrict__ Z) {
    __shared__ __align__(16) float Alds[64 * GLDS];
    __shared__ __align__(16) float Bw[32 * 64];

    const int ib = blockIdx.y * 64;
    const int nb = blockIdx.x * 64;
    const int tid  = threadIdx.x;
    const int lane = tid & 31;
    const int wv   = tid >> 5;
    const int sr   = wv & 3;
    const int sc0  = (wv >> 2) << 1;
    const int lr   = lane & 15;
    const int kq   = (lane >> 4) << 1;

    v8f acc0 = {}; v8f acc1 = {};

    for (int k0 = 0; k0 < DIM; k0 += 32) {
        __syncthreads();
        for (int f = tid; f < 512; f += 256) {          // A: 64 rows x 32 k
            const int r = f >> 3, c = (f & 7) << 2;
            stage_b128(&Alds[r * GLDS + c], &X[(size_t)(ib + r) * DIM + k0 + c]);
        }
        for (int f = tid; f < 512; f += 256) {          // B: 32 k x 64 n
            const int r = f >> 4, c = (f & 15) << 2;
            stage_b128(&Bw[r * 64 + c], &We[(size_t)(k0 + r) * LATENT + nb + c]);
        }
        stage_fence();
        __syncthreads();
        const float* Ar = &Alds[(sr * 16 + lr) * GLDS + kq];
        #pragma unroll
        for (int kk = 0; kk < 32; kk += 4) {
            const int kr = kk + kq;
            v2f a, b0, b1;
            a.x  = Ar[kk];                           a.y  = Ar[kk + 1];
            b0.x = Bw[kr * 64 + sc0 * 16 + lr];      b0.y = Bw[(kr + 1) * 64 + sc0 * 16 + lr];
            b1.x = Bw[kr * 64 + sc0 * 16 + 16 + lr]; b1.y = Bw[(kr + 1) * 64 + sc0 * 16 + 16 + lr];
            acc0 = wmma_f32(a, b0, acc0);
            acc1 = wmma_f32(a, b1, acc1);
        }
    }

    const int i0 = ib + sr * 16 + ((lane >> 4) << 3);
    const int j0 = nb + sc0 * 16 + lr;
    const int j1 = j0 + 16;
    const float b0v = be[j0], b1v = be[j1];
    #pragma unroll
    for (int e = 0; e < 8; ++e) {
        const int ii = i0 + e;
        Z[(size_t)ii * LATENT + j0] = acc0[e] + b0v;
        Z[(size_t)ii * LATENT + j1] = acc1[e] + b1v;
    }
}

// ---------------------------------------------------------------------------
// Decoder + reconstruction loss partials:
//   out = Z @ Wd + bd ; part[block] = sum (out - X)^2 over the 64x64 tile.
// ---------------------------------------------------------------------------
__global__ __launch_bounds__(256)
void decode_recon_kernel(const float* __restrict__ Z, const float* __restrict__ Wd,
                         const float* __restrict__ bd, const float* __restrict__ X,
                         float* __restrict__ recon_part) {
    __shared__ __align__(16) float Alds[64 * GLDS];
    __shared__ __align__(16) float Bw[32 * 64];
    __shared__ float red[8];

    const int ib = blockIdx.y * 64;
    const int nb = blockIdx.x * 64;
    const int tid  = threadIdx.x;
    const int lane = tid & 31;
    const int wv   = tid >> 5;
    const int sr   = wv & 3;
    const int sc0  = (wv >> 2) << 1;
    const int lr   = lane & 15;
    const int kq   = (lane >> 4) << 1;

    v8f acc0 = {}; v8f acc1 = {};

    for (int k0 = 0; k0 < LATENT; k0 += 32) {
        __syncthreads();
        for (int f = tid; f < 512; f += 256) {          // A: 64 rows of Z x 32 k
            const int r = f >> 3, c = (f & 7) << 2;
            stage_b128(&Alds[r * GLDS + c], &Z[(size_t)(ib + r) * LATENT + k0 + c]);
        }
        for (int f = tid; f < 512; f += 256) {          // B: 32 k x 64 n of Wd
            const int r = f >> 4, c = (f & 15) << 2;
            stage_b128(&Bw[r * 64 + c], &Wd[(size_t)(k0 + r) * DIM + nb + c]);
        }
        stage_fence();
        __syncthreads();
        const float* Ar = &Alds[(sr * 16 + lr) * GLDS + kq];
        #pragma unroll
        for (int kk = 0; kk < 32; kk += 4) {
            const int kr = kk + kq;
            v2f a, b0, b1;
            a.x  = Ar[kk];                           a.y  = Ar[kk + 1];
            b0.x = Bw[kr * 64 + sc0 * 16 + lr];      b0.y = Bw[(kr + 1) * 64 + sc0 * 16 + lr];
            b1.x = Bw[kr * 64 + sc0 * 16 + 16 + lr]; b1.y = Bw[(kr + 1) * 64 + sc0 * 16 + 16 + lr];
            acc0 = wmma_f32(a, b0, acc0);
            acc1 = wmma_f32(a, b1, acc1);
        }
    }

    const int i0 = ib + sr * 16 + ((lane >> 4) << 3);
    const int j0 = nb + sc0 * 16 + lr;
    const int j1 = j0 + 16;
    const float b0v = bd[j0], b1v = bd[j1];
    float s = 0.f;
    #pragma unroll
    for (int e = 0; e < 8; ++e) {
        const int ii = i0 + e;
        float e0 = acc0[e] + b0v - X[(size_t)ii * DIM + j0];
        float e1 = acc1[e] + b1v - X[(size_t)ii * DIM + j1];
        s += e0 * e0 + e1 * e1;
    }
    s = wave_sum(s);
    if (lane == 0) red[wv] = s;
    __syncthreads();
    if (tid == 0) {
        float t = 0.f;
        #pragma unroll
        for (int w = 0; w < 8; ++w) t += red[w];
        recon_part[blockIdx.y * gridDim.x + blockIdx.x] = t;
    }
}

// ---------------------------------------------------------------------------
// Prim's MST (== 0-dim persistence pairing).  One 1024-thread block (32 waves),
// 4 columns / thread, wave32 shuffle argmin + 32-entry LDS reduce per step.
// ---------------------------------------------------------------------------
__global__ __launch_bounds__(1024)
void mst_kernel(const float* __restrict__ dist, int2* __restrict__ pairs) {
    const int n = NPTS;
    const int tid  = threadIdx.x;
    const int lane = tid & 31;
    const int wv   = tid >> 5;
    __shared__ float swv[32];
    __shared__ int   swi[32];
    __shared__ int   s_uv[2];   // [0]=u, [1]=v

    float mind[4];
    int   msrc[4];
    bool  intree[4];
    #pragma unroll
    for (int c = 0; c < 4; ++c) {
        const int col = tid + (c << 10);
        mind[c]   = dist[col];          // distances from node 0
        msrc[c]   = 0;
        intree[c] = false;
    }
    if (tid == 0) { intree[0] = true; mind[0] = __builtin_inff(); }

    for (int step = 0; step < n - 1; ++step) {
        float bv = mind[0]; int bi = tid;
        #pragma unroll
        for (int c = 1; c < 4; ++c) {
            const int col = tid + (c << 10);
            if (mind[c] < bv) { bv = mind[c]; bi = col; }
        }
        #pragma unroll
        for (int off = 16; off > 0; off >>= 1) {
            float ov = __shfl_down(bv, off);
            int   oi = __shfl_down(bi, off);
            if (ov < bv) { bv = ov; bi = oi; }
        }
        if (lane == 0) { swv[wv] = bv; swi[wv] = bi; }
        __syncthreads();
        if (wv == 0) {                   // 32 wave partials -> global argmin
            float v = swv[lane];
            int   i = swi[lane];
            #pragma unroll
            for (int off = 16; off > 0; off >>= 1) {
                float ov = __shfl_down(v, off);
                int   oi = __shfl_down(i, off);
                if (ov < v) { v = ov; i = oi; }
            }
            if (lane == 0) s_uv[1] = i;
        }
        __syncthreads();
        const int v = s_uv[1];
        const int c = v >> 10;
        if ((v & 1023) == tid) {         // owner publishes u, marks in-tree
            s_uv[0] = msrc[c];
            intree[c] = true;
            mind[c] = __builtin_inff();
        }
        __syncthreads();
        if (tid == 0) pairs[step] = make_int2(s_uv[0], v);
        const float* row = dist + (size_t)v * n;
        #pragma unroll
        for (int cc = 0; cc < 4; ++cc) {
            const int col = tid + (cc << 10);
            const float dv = row[col];
            if (!intree[cc] && dv < mind[cc]) { mind[cc] = dv; msrc[cc] = v; }
        }
        __syncthreads();
    }
}

// ---------------------------------------------------------------------------
// Signature loss partials: part[b] = sum over pairs of 0.5*(dx[p]-dz[p])^2
// ---------------------------------------------------------------------------
__global__ __launch_bounds__(256)
void sig_loss_kernel(const float* __restrict__ dx, const float* __restrict__ dz,
                     const int2* __restrict__ pairs, int m, float* __restrict__ part) {
    __shared__ float red[8];
    const int t = blockIdx.x * 256 + threadIdx.x;
    float s = 0.f;
    if (t < m) {
        const int2 p = pairs[t];
        const size_t off = (size_t)p.x * NPTS + p.y;
        const float d = dx[off] - dz[off];
        s = 0.5f * d * d;
    }
    s = wave_sum(s);
    const int lane = threadIdx.x & 31, wv = threadIdx.x >> 5;
    if (lane == 0) red[wv] = s;
    __syncthreads();
    if (threadIdx.x == 0) {
        float t2 = 0.f;
        #pragma unroll
        for (int w = 0; w < 8; ++w) t2 += red[w];
        part[blockIdx.x] = t2;
    }
}

// ---------------------------------------------------------------------------
// Deterministic final reduction: out = mean(recon) + topo
// ---------------------------------------------------------------------------
__global__ __launch_bounds__(256)
void finalize_kernel(const float* __restrict__ recon_part,
                     const float* __restrict__ sx, const float* __restrict__ sz,
                     float* __restrict__ out) {
    __shared__ float red[8];
    const int tid = threadIdx.x;
    float s = 0.f;
    for (int i = tid; i < 8192; i += 256) s += recon_part[i];
    s = wave_sum(s);
    const int lane = tid & 31, wv = tid >> 5;
    if (lane == 0) red[wv] = s;
    __syncthreads();
    if (tid == 0) {
        float recon = 0.f;
        #pragma unroll
        for (int w = 0; w < 8; ++w) recon += red[w];
        float topo = 0.f;
        for (int i = 0; i < 16; ++i) topo += sx[i] + sz[i];
        out[0] = recon / ((float)NPTS * (float)DIM) + topo;
    }
}

// ---------------------------------------------------------------------------
// Host-side launcher (graph-capture safe: only kernel launches on `stream`).
// ---------------------------------------------------------------------------
extern "C" void kernel_launch(void* const* d_in, const int* in_sizes, int n_in,
                              void* d_out, int out_size, void* d_ws, size_t ws_size,
                              hipStream_t stream) {
    (void)in_sizes; (void)n_in; (void)out_size; (void)ws_size;

    const float* x  = (const float*)d_in[0];   // [4096, 8192]
    const float* We = (const float*)d_in[1];   // [8192, 128]
    const float* be = (const float*)d_in[2];   // [128]
    const float* Wd = (const float*)d_in[3];   // [128, 8192]
    const float* bd = (const float*)d_in[4];   // [8192]
    float* out = (float*)d_out;                // scalar loss

    // Workspace carve-up (~136.5 MB)
    char* w = (char*)d_ws;
    float* dx   = (float*)(w + 0);             // 4096*4096*4 = 64 MB
    float* dz   = (float*)(w + 67108864ull);   // 64 MB
    float* z    = (float*)(w + 134217728ull);  // 4096*128*4 = 2 MB
    float* sqx  = (float*)(w + 136314880ull);  // 16 KB
    float* sqz  = (float*)(w + 136331264ull);  // 16 KB
    int2*  px   = (int2*) (w + 136347648ull);  // 4095*8 B
    int2*  pz   = (int2*) (w + 136380416ull);  // 4095*8 B
    float* rpar = (float*)(w + 136413184ull);  // 8192*4 = 32 KB recon partials
    float* sxp  = (float*)(w + 136445952ull);  // 16 floats
    float* szp  = (float*)(w + 136446016ull);  // 16 floats

    // 1) ||x_i||^2 and dx = pairwise_dist(x)  (dominant WMMA work)
    rowsq_kernel<<<NPTS, 256, 0, stream>>>(x, sqx, DIM);
    gram_dist_kernel<<<dim3(NPTS / 64, NPTS / 64), 256, 0, stream>>>(x, sqx, dx, NPTS, DIM);

    // 2) z = x @ We + be ; dz = pairwise_dist(z)
    encode_kernel<<<dim3(LATENT / 64, NPTS / 64), 256, 0, stream>>>(x, We, be, z);
    rowsq_kernel<<<NPTS, 256, 0, stream>>>(z, sqz, LATENT);
    gram_dist_kernel<<<dim3(NPTS / 64, NPTS / 64), 256, 0, stream>>>(z, sqz, dz, NPTS, LATENT);

    // 3) 0-dim persistence pairings (Prim MST) for both spaces
    mst_kernel<<<1, 1024, 0, stream>>>(dx, px);
    mst_kernel<<<1, 1024, 0, stream>>>(dz, pz);

    // 4) topological signature loss partials
    sig_loss_kernel<<<16, 256, 0, stream>>>(dx, dz, px, NPTS - 1, sxp);
    sig_loss_kernel<<<16, 256, 0, stream>>>(dx, dz, pz, NPTS - 1, szp);

    // 5) decode + reconstruction MSE partials
    decode_recon_kernel<<<dim3(DIM / 64, NPTS / 64), 256, 0, stream>>>(z, Wd, bd, x, rpar);

    // 6) deterministic combine
    finalize_kernel<<<1, 256, 0, stream>>>(rpar, sxp, szp, out);
}
